// DeformableTransformerEncoderLayer_62586263437932
// MI455X (gfx1250) — matrix operand order repaired and checked by reading, not compiled, not run on hardware
//
#include <hip/hip_runtime.h>
#include <hip/hip_bf16.h>
#include <math.h>

typedef _Float16 v16h __attribute__((ext_vector_type(16)));
typedef _Float16 v8h  __attribute__((ext_vector_type(8)));
typedef float    v8f  __attribute__((ext_vector_type(8)));
typedef int      v4i  __attribute__((ext_vector_type(4)));

// Pointer types matching the async-copy builtin's expected parameters:
// arg0: int4 addrspace(1)* (global source), arg1: int4 addrspace(3)* (LDS dst)
typedef __attribute__((address_space(1))) v4i* as1_v4i_ptr;
typedef __attribute__((address_space(3))) v4i* as3_v4i_ptr;

#define NB    4
#define NLEN  5440
#define NC    256
#define NH    8
#define NDH   32
#define NL    4
#define NP    4
#define NDFF  1024
#define MROWS (NB * NLEN)

// CDNA5 async global->LDS path (ASYNCcnt-tracked). Gated so the source still
// compiles (with a synchronous LDS-staging fallback) if the toolchain lacks
// the builtin.
#if defined(__has_builtin)
#if __has_builtin(__builtin_amdgcn_global_load_async_to_lds_b128)
#define HAVE_ASYNC_LDS 1
#endif
#if __has_builtin(__builtin_amdgcn_s_wait_asynccnt)
#define HAVE_WAIT_ASYNC_BUILTIN 1
#endif
#endif

__device__ __forceinline__ void wait_async_all()
{
#ifdef HAVE_ASYNC_LDS
#ifdef HAVE_WAIT_ASYNC_BUILTIN
  __builtin_amdgcn_s_wait_asynccnt(0);
#else
  asm volatile("s_wait_asynccnt 0x0" ::: "memory");
#endif
#endif
}

// ---------------------------------------------------------------------------
// pack: q16 = f16(src + pos), s16 = f16(src)
// ---------------------------------------------------------------------------
__global__ __launch_bounds__(256) void pack_q_kernel(
    const float* __restrict__ src, const float* __restrict__ pos,
    _Float16* __restrict__ q16, _Float16* __restrict__ s16, int n)
{
  int i = blockIdx.x * 256 + threadIdx.x;
  if (i < n) {
    float s = src[i];
    q16[i] = (_Float16)(s + pos[i]);
    s16[i] = (_Float16)s;
  }
}

// ---------------------------------------------------------------------------
// transpose + convert: f32 W[K][N] (row-major) -> f16 Wt[N][K] (row-major)
// ---------------------------------------------------------------------------
__global__ __launch_bounds__(256) void transpose_f16_kernel(
    const float* __restrict__ w, _Float16* __restrict__ wt, int K, int N)
{
  __shared__ _Float16 tile[32][33];
  int tx = threadIdx.x & 31;
  int ty = threadIdx.x >> 5;          // 0..7
  int n0 = blockIdx.x * 32;
  int k0 = blockIdx.y * 32;
#pragma unroll
  for (int i = 0; i < 4; ++i) {
    int k = k0 + ty + i * 8;
    tile[ty + i * 8][tx] = (_Float16)w[(size_t)k * N + (n0 + tx)];
  }
  __syncthreads();
#pragma unroll
  for (int i = 0; i < 4; ++i) {
    int n = n0 + ty + i * 8;
    wt[(size_t)n * K + (k0 + tx)] = tile[tx][ty + i * 8];
  }
}

// ---------------------------------------------------------------------------
// WMMA fragment loaders (wave32, 16-bit operands)
// A 16x32 layout: lane<16 -> row M=lane, K = [k0..k0+15]
//                 lane>=16 -> same row, K = [k0+16..k0+31]
// ---------------------------------------------------------------------------
__device__ __forceinline__ v16h load_fragA(const _Float16* __restrict__ base,
                                           int ld, int row0, int k0)
{
  int lane = threadIdx.x & 31;
  const _Float16* p =
      base + (size_t)(row0 + (lane & 15)) * ld + k0 + (lane >> 4) * 16;
  v8h lo = *(const v8h*)p;
  v8h hi = *(const v8h*)(p + 8);
  return __builtin_shufflevector(lo, hi, 0, 1, 2, 3, 4, 5, 6, 7, 8, 9, 10, 11,
                                 12, 13, 14, 15);
}

// B 32x16 layout: lane<16 -> col N=lane, K = [k0..k0+15];
//                 lane>=16 -> K = [k0+16..k0+31]. p must point at (row, k0 + 16*(lane/16)).
__device__ __forceinline__ v16h make_fragB(const _Float16* p)
{
  v8h lo = *(const v8h*)p;
  v8h hi = *(const v8h*)(p + 8);
  return __builtin_shufflevector(lo, hi, 0, 1, 2, 3, 4, 5, 6, 7, 8, 9, 10, 11,
                                 12, 13, 14, 15);
}

// ---------------------------------------------------------------------------
// Generic f16xf16->f32 WMMA GEMM: C[m,n] = sum_k A[m,k]*Bt[n,k] + bias[n]
// Block = 256 threads = 8 waves (4 in M x 2 in N); wave computes 32x32.
// B tile (64 N x 64 K halfs per stage) is staged in LDS, double-buffered,
// filled with CDNA5 async global->LDS copies overlapped with WMMA compute.
// A is streamed straight from global (each chunk used once per wave-pair).
// flags: bit0 = ReLU, bit1 = store f16 (else f32).
// Requires M%128==0, N%64==0, K%64==0 (true for all calls here).
// ---------------------------------------------------------------------------
#define KSTAGE 64
#define BPAD   72   // 64 + 8 halfs row padding (16B-aligned rows: 144 bytes)

__global__ __launch_bounds__(256) void gemm_wmma_kernel(
    const _Float16* __restrict__ A, const _Float16* __restrict__ Bt,
    const float* __restrict__ bias, float* __restrict__ C32,
    _Float16* __restrict__ C16, int Mdim, int Ndim, int Kdim, int flags)
{
  __shared__ _Float16 ldsB[2][64][BPAD];

  const int wave     = threadIdx.x >> 5;
  const int lane     = threadIdx.x & 31;
  const int rowBase  = blockIdx.x * 128 + (wave & 3) * 32;
  const int colBlock = blockIdx.y * 64;            // block's N origin
  const int rowLocN  = (wave >> 2) * 32;           // wave's N offset in tile

  // --- stage copy: 64 rows x 64 halfs = 512 x 16B chunks, 2 per thread ----
  auto copy_stage = [&](int buf, int ks) {
#pragma unroll
    for (int c = threadIdx.x; c < 512; c += 256) {
      int row = c >> 3;
      int kc  = (c & 7) * 8;
      const _Float16* g = Bt + (size_t)(colBlock + row) * Kdim + ks + kc;
      _Float16* l = &ldsB[buf][row][kc];
#ifdef HAVE_ASYNC_LDS
      __builtin_amdgcn_global_load_async_to_lds_b128(
          (as1_v4i_ptr)g, (as3_v4i_ptr)l, 0, 0);
#else
      *(v8h*)l = *(const v8h*)g;
#endif
    }
  };

  v8f acc00 = {}, acc01 = {}, acc10 = {}, acc11 = {};

  int buf = 0;
  copy_stage(0, 0);
  wait_async_all();
  __syncthreads();

  for (int ks = 0; ks < Kdim; ks += KSTAGE) {
    if (ks + KSTAGE < Kdim) copy_stage(buf ^ 1, ks + KSTAGE);   // overlap

#pragma unroll
    for (int kk = 0; kk < KSTAGE; kk += 32) {
      const int k0 = ks + kk;
      __builtin_prefetch(
          (const void*)(A + (size_t)(rowBase + (lane & 15)) * Kdim + k0 + 64),
          0, 0);
      v16h a0 = load_fragA(A, Kdim, rowBase, k0);
      v16h a1 = load_fragA(A, Kdim, rowBase + 16, k0);
      const int kl = kk + (lane >> 4) * 16;
      v16h b0 = make_fragB(&ldsB[buf][rowLocN + (lane & 15)][kl]);
      v16h b1 = make_fragB(&ldsB[buf][rowLocN + 16 + (lane & 15)][kl]);
      acc00 = __builtin_amdgcn_wmma_f32_16x16x32_f16(false, a0, false, b0,
                                                     (short)0, acc00, false,
                                                     false);
      acc01 = __builtin_amdgcn_wmma_f32_16x16x32_f16(false, a0, false, b1,
                                                     (short)0, acc01, false,
                                                     false);
      acc10 = __builtin_amdgcn_wmma_f32_16x16x32_f16(false, a1, false, b0,
                                                     (short)0, acc10, false,
                                                     false);
      acc11 = __builtin_amdgcn_wmma_f32_16x16x32_f16(false, a1, false, b1,
                                                     (short)0, acc11, false,
                                                     false);
    }

    wait_async_all();     // next-stage async copies have landed
    __syncthreads();      // all waves done reading current buffer
    buf ^= 1;
  }

  const int relu   = flags & 1;
  const int f16out = flags & 2;
  const int nlo    = lane & 15;
  const int mhalf  = (lane >> 4) * 8;
#pragma unroll
  for (int i = 0; i < 2; ++i) {
#pragma unroll
    for (int j = 0; j < 2; ++j) {
      v8f acc = (i == 0) ? (j == 0 ? acc00 : acc01) : (j == 0 ? acc10 : acc11);
      int n    = colBlock + rowLocN + j * 16 + nlo;
      float bv = bias ? bias[n] : 0.f;
#pragma unroll
      for (int r = 0; r < 8; ++r) {
        int m   = rowBase + i * 16 + mhalf + r;
        float v = acc[r] + bv;
        if (relu) v = fmaxf(v, 0.f);
        size_t idx = (size_t)m * Ndim + n;
        if (f16out) C16[idx] = (_Float16)v;
        else        C32[idx] = v;
      }
    }
  }
}

// ---------------------------------------------------------------------------
// softmax over the 16 contiguous (L*P) logits per (m,h); in place.
// ---------------------------------------------------------------------------
__global__ __launch_bounds__(256) void softmax16_kernel(float* __restrict__ buf,
                                                        int rows)
{
  int r = blockIdx.x * 256 + threadIdx.x;
  if (r >= rows) return;
  float* p = buf + (size_t)r * 16;
  float v[16];
#pragma unroll
  for (int i = 0; i < 16; ++i) v[i] = p[i];
  float mx = v[0];
#pragma unroll
  for (int i = 1; i < 16; ++i) mx = fmaxf(mx, v[i]);
  float s = 0.f;
#pragma unroll
  for (int i = 0; i < 16; ++i) { v[i] = __expf(v[i] - mx); s += v[i]; }
  float inv = 1.f / s;
#pragma unroll
  for (int i = 0; i < 16; ++i) p[i] = v[i] * inv;
}

// ---------------------------------------------------------------------------
// MSDA core: one wave32 per (m, h), lane = channel dh (32 channels -> every
// corner gather is one contiguous 128B line). Writes f16 for the W_out GEMM.
// ---------------------------------------------------------------------------
__global__ __launch_bounds__(256) void msda_kernel(
    const float* __restrict__ off, const float* __restrict__ refp,
    const float* __restrict__ aw, const float* __restrict__ value,
    _Float16* __restrict__ out16)
{
  const int wave = threadIdx.x >> 5;
  const int lane = threadIdx.x & 31;
  int g = blockIdx.x * 8 + wave;
  if (g >= MROWS * NH) return;
  int m = g >> 3;     // flattened (b, lq)
  int h = g & 7;
  int b = m / NLEN;

  const float* offm = off + (size_t)m * NC + h * (NL * NP * 2);   // [l][p][2]
  const float* awm  = aw + (size_t)m * (NH * NL * NP) + h * (NL * NP);
  const float* refm = refp + (size_t)m * (NL * 2);
  const float* vb   = value + (size_t)b * NLEN * NC + h * NDH + lane;

  const int Hls[4] = {64, 32, 16, 8};
  const int Wls[4] = {64, 32, 16, 8};
  const int lsi[4] = {0, 4096, 5120, 5376};

  float acc = 0.f;
#pragma unroll
  for (int l = 0; l < 4; ++l) {
    const int Wl = Wls[l], Hl = Hls[l];
    const float rx = refm[l * 2 + 0] * (float)Wl;
    const float ry = refm[l * 2 + 1] * (float)Hl;
    const float* vlev = vb + (size_t)lsi[l] * NC;
#pragma unroll
    for (int p = 0; p < 4; ++p) {
      // loc*W - 0.5 == ref*W + off - 0.5  (off already divided by norm=W)
      float px = rx + offm[(l * 4 + p) * 2 + 0] - 0.5f;
      float py = ry + offm[(l * 4 + p) * 2 + 1] - 0.5f;
      float a  = awm[l * 4 + p];
      float x0f = floorf(px), y0f = floorf(py);
      float wx1 = px - x0f, wy1 = py - y0f;
      float wx0 = 1.f - wx1, wy0 = 1.f - wy1;
      int x0 = (int)x0f, y0 = (int)y0f;
#pragma unroll
      for (int c = 0; c < 4; ++c) {
        int ix = x0 + (c & 1);
        int iy = y0 + ((c >> 1) & 1);
        float w = ((c & 1) ? wx1 : wx0) * ((c & 2) ? wy1 : wy0);
        bool valid = (ix >= 0) && (ix < Wl) && (iy >= 0) && (iy < Hl);
        int cx = ix < 0 ? 0 : (ix > Wl - 1 ? Wl - 1 : ix);
        int cy = iy < 0 ? 0 : (iy > Hl - 1 ? Hl - 1 : iy);
        float gv = vlev[(size_t)(cy * Wl + cx) * NC];
        acc += gv * (valid ? w * a : 0.f);
      }
    }
  }
  out16[(size_t)m * NC + h * NDH + lane] = (_Float16)acc;
}

// ---------------------------------------------------------------------------
// LayerNorm over 256 channels of (a + b): one wave32 per row, shfl_xor
// reduction. Writes f32 out (residual source) and optional f16 (GEMM A).
// ---------------------------------------------------------------------------
__global__ __launch_bounds__(256) void ln_kernel(
    const float* __restrict__ a, const float* __restrict__ b,
    const float* __restrict__ gamma, const float* __restrict__ beta,
    float* __restrict__ out32, _Float16* __restrict__ out16, int rows)
{
  const int wave = threadIdx.x >> 5;
  const int lane = threadIdx.x & 31;
  int r = blockIdx.x * 8 + wave;
  if (r >= rows) return;
  const float* pa = a + (size_t)r * NC;
  const float* pb = b + (size_t)r * NC;
  float v[8];
  float s = 0.f, ss = 0.f;
#pragma unroll
  for (int i = 0; i < 8; ++i) {
    int c = lane + i * 32;
    float x = pa[c] + pb[c];
    v[i] = x; s += x; ss += x * x;
  }
#pragma unroll
  for (int msk = 16; msk >= 1; msk >>= 1) {
    s  += __shfl_xor(s, msk, 32);
    ss += __shfl_xor(ss, msk, 32);
  }
  const float mean = s * (1.f / NC);
  const float var  = ss * (1.f / NC) - mean * mean;
  const float inv  = rsqrtf(var + 1e-5f);
#pragma unroll
  for (int i = 0; i < 8; ++i) {
    int c = lane + i * 32;
    float y = (v[i] - mean) * inv * gamma[c] + beta[c];
    if (out32) out32[(size_t)r * NC + c] = y;
    if (out16) out16[(size_t)r * NC + c] = (_Float16)y;
  }
}

// ---------------------------------------------------------------------------
extern "C" void kernel_launch(void* const* d_in, const int* in_sizes, int n_in,
                              void* d_out, int out_size, void* d_ws,
                              size_t ws_size, hipStream_t stream)
{
  (void)in_sizes; (void)n_in; (void)out_size; (void)ws_size;

  const float* src   = (const float*)d_in[0];
  const float* pos   = (const float*)d_in[1];
  const float* refp  = (const float*)d_in[2];
  // d_in[3] spatial_shapes, d_in[4] level_start_index, d_in[5] valid_ratios:
  // constants for this problem; hardcoded in the kernels.
  const float* w_off  = (const float*)d_in[6];
  const float* b_off  = (const float*)d_in[7];
  const float* w_attn = (const float*)d_in[8];
  const float* b_attn = (const float*)d_in[9];
  const float* w_val  = (const float*)d_in[10];
  const float* b_val  = (const float*)d_in[11];
  const float* w_out  = (const float*)d_in[12];
  const float* b_out  = (const float*)d_in[13];
  const float* g1     = (const float*)d_in[14];
  const float* be1    = (const float*)d_in[15];
  const float* w1     = (const float*)d_in[16];
  const float* b1     = (const float*)d_in[17];
  const float* w2     = (const float*)d_in[18];
  const float* b2     = (const float*)d_in[19];
  const float* g2     = (const float*)d_in[20];
  const float* be2    = (const float*)d_in[21];

  char* ws = (char*)d_ws;
  size_t cursor = 0;
  auto alloc = [&](size_t bytes) {
    size_t o = cursor;
    cursor = (cursor + bytes + 255) & ~(size_t)255;
    return o;
  };

  const size_t M = MROWS;
  _Float16* q16    = (_Float16*)(ws + alloc(M * NC * 2));    // reused: msda16
  _Float16* s16    = (_Float16*)(ws + alloc(M * NC * 2));    // reused: x16
  _Float16* woffT  = (_Float16*)(ws + alloc(256 * 256 * 2));
  _Float16* wattnT = (_Float16*)(ws + alloc(128 * 256 * 2));
  _Float16* wvalT  = (_Float16*)(ws + alloc(256 * 256 * 2));
  _Float16* woutT  = (_Float16*)(ws + alloc(256 * 256 * 2));
  _Float16* w1T    = (_Float16*)(ws + alloc(1024 * 256 * 2));
  _Float16* w2T    = (_Float16*)(ws + alloc(256 * 1024 * 2));
  float* off32     = (float*)(ws + alloc(M * NC * 4));       // reused: attn_out
  float* attn32    = (float*)(ws + alloc(M * 128 * 4));
  float* val32     = (float*)(ws + alloc(M * NC * 4));       // reused: ffn2
  float* x32       = (float*)(ws + alloc(M * NC * 4));
  _Float16* h1     = (_Float16*)(ws + alloc(M * NDFF * 2));

  // 1) pack activations to f16
  pack_q_kernel<<<(unsigned)((M * NC) / 256), 256, 0, stream>>>(
      src, pos, q16, s16, (int)(M * NC));

  // 2) transpose + f16-convert all weights to [N][K]
  transpose_f16_kernel<<<dim3(8, 8),  256, 0, stream>>>(w_off,  woffT,  256, 256);
  transpose_f16_kernel<<<dim3(4, 8),  256, 0, stream>>>(w_attn, wattnT, 256, 128);
  transpose_f16_kernel<<<dim3(8, 8),  256, 0, stream>>>(w_val,  wvalT,  256, 256);
  transpose_f16_kernel<<<dim3(8, 8),  256, 0, stream>>>(w_out,  woutT,  256, 256);
  transpose_f16_kernel<<<dim3(32, 8), 256, 0, stream>>>(w1,     w1T,    256, 1024);
  transpose_f16_kernel<<<dim3(8, 32), 256, 0, stream>>>(w2,     w2T,    1024, 256);

  const dim3 blk(256);
  // 3) sampling offsets: off = q @ W_off + b_off        [M, 256]
  gemm_wmma_kernel<<<dim3((unsigned)(M / 128), 4), blk, 0, stream>>>(
      q16, woffT, b_off, off32, nullptr, (int)M, 256, 256, 0);
  // 4) attention logits: q @ W_attn + b_attn            [M, 128]
  gemm_wmma_kernel<<<dim3((unsigned)(M / 128), 2), blk, 0, stream>>>(
      q16, wattnT, b_attn, attn32, nullptr, (int)M, 128, 256, 0);
  // 5) softmax over L*P = 16 per (m, h)
  softmax16_kernel<<<(unsigned)((M * NH) / 256), 256, 0, stream>>>(
      attn32, (int)(M * NH));
  // 6) value projection: src @ W_val + b_val            [M, 256]
  gemm_wmma_kernel<<<dim3((unsigned)(M / 128), 4), blk, 0, stream>>>(
      s16, wvalT, b_val, val32, nullptr, (int)M, 256, 256, 0);

  // 7) deformable attention gather (writes f16, reuses q16)
  _Float16* msda16 = q16;
  msda_kernel<<<(unsigned)((M * NH) / 8), 256, 0, stream>>>(
      off32, refp, attn32, val32, msda16);

  // 8) output projection (reuses off32 as attn_out)
  float* attnout32 = off32;
  gemm_wmma_kernel<<<dim3((unsigned)(M / 128), 4), blk, 0, stream>>>(
      msda16, woutT, b_out, attnout32, nullptr, (int)M, 256, 256, 0);

  // 9) x = LN(src + attn_out); emits f32 (residual) + f16 (FFN A)
  _Float16* x16 = s16;
  ln_kernel<<<(unsigned)(M / 8), 256, 0, stream>>>(
      src, attnout32, g1, be1, x32, x16, (int)M);

  // 10) FFN up: relu(x @ W1 + b1) stored directly as f16 [M, 1024]
  gemm_wmma_kernel<<<dim3((unsigned)(M / 128), 16), blk, 0, stream>>>(
      x16, w1T, b1, nullptr, h1, (int)M, 1024, 256, /*relu|f16*/ 3);

  // 11) FFN down: h1 @ W2 + b2 (reuses val32 as ffn2)
  float* ffn2 = val32;
  gemm_wmma_kernel<<<dim3((unsigned)(M / 128), 4), blk, 0, stream>>>(
      h1, w2T, b2, ffn2, nullptr, (int)M, 256, 1024, 0);

  // 12) out = LN(x + ffn2)
  ln_kernel<<<(unsigned)(M / 8), 256, 0, stream>>>(
      x32, ffn2, g2, be2, (float*)d_out, nullptr, (int)M);
}
